// MultiHeadAttention_v2M_79405355369265
// MI455X (gfx1250) — matrix operand order repaired and checked by reading, not compiled, not run on hardware
//
#include <hip/hip_runtime.h>
#include <cstdint>
#include <cstddef>

// ---------------------------------------------------------------------------
// MHA forward for MI455X (gfx1250, wave32) via v_wmma_f32_16x16x32_bf16.
// B=2, T=2048, D=1024, H=16, hd=64.
// 32x64 register tiles: every B-fragment feeds 2 WMMAs (2 A row-tiles).
// ---------------------------------------------------------------------------

typedef __bf16 bf16;
typedef __attribute__((ext_vector_type(16))) __bf16 v16bf;
typedef __attribute__((ext_vector_type(8)))  __bf16 v8bf;
typedef __attribute__((ext_vector_type(8)))  float  v8f;

static constexpr int Bb  = 2;
static constexpr int T   = 2048;
static constexpr int D   = 1024;
static constexpr int H   = 16;
static constexpr int HD  = 64;
static constexpr int M   = Bb * T;      // 4096 rows

union V16U { v16bf v; v8bf h[2]; };

// A-fragment (16x32 bf16): lane L -> row L%16; K = {8*(L/16)+0..7, +16..23}
__device__ __forceinline__ v16bf load_afrag(const bf16* __restrict__ base, int ld,
                                            int row0, int k0, int lane) {
    int r  = row0 + (lane & 15);
    int kb = k0 + ((lane >> 4) << 3);
    const bf16* p = base + (size_t)r * ld + kb;
    V16U u;
    u.h[0] = *(const v8bf*)(p);
    u.h[1] = *(const v8bf*)(p + 16);
    return u.v;
}

// B-fragment (32x16 bf16) from Bt stored N-major [N][K]:
// lane L -> col L%16; K = {8*(L/16)+0..7, +16..23}
__device__ __forceinline__ v16bf load_bfrag(const bf16* __restrict__ bt, int ld,
                                            int n0, int k0, int lane) {
    int n  = n0 + (lane & 15);
    int kb = k0 + ((lane >> 4) << 3);
    const bf16* p = bt + (size_t)n * ld + kb;
    V16U u;
    u.h[0] = *(const v8bf*)(p);
    u.h[1] = *(const v8bf*)(p + 16);
    return u.v;
}

__device__ __forceinline__ v8f wmma_bf16(v16bf a, v16bf b, v8f c) {
    return __builtin_amdgcn_wmma_f32_16x16x32_bf16(false, a, false, b,
                                                   (short)0, c, false, false);
}

__device__ __forceinline__ float rmax16(float v) {
    #pragma unroll
    for (int m = 1; m < 16; m <<= 1) v = fmaxf(v, __shfl_xor(v, m, 32));
    return v;
}
__device__ __forceinline__ float rsum16(float v) {
    #pragma unroll
    for (int m = 1; m < 16; m <<= 1) v += __shfl_xor(v, m, 32);
    return v;
}

// ---------------------------------------------------------------------------
// Stage 0: precision conversion / weight transposition
// ---------------------------------------------------------------------------
__global__ void cvt_f32_to_bf16(const float* __restrict__ in, bf16* __restrict__ out, int n) {
    int i = blockIdx.x * blockDim.x + threadIdx.x;
    if (i < n) out[i] = (bf16)in[i];
}

// in: [rows][cols] f32  ->  out: [cols][rows] bf16
__global__ void transpose_to_bf16(const float* __restrict__ in, bf16* __restrict__ out,
                                  int rows, int cols) {
    int i = blockIdx.x * blockDim.x + threadIdx.x;
    if (i < rows * cols) {
        int cc = i / rows, rr = i - cc * rows;
        out[(size_t)cc * rows + rr] = (bf16)in[(size_t)rr * cols + cc];
    }
}

// ---------------------------------------------------------------------------
// Stage 1: QKV projection + RoPE + head scatter.
// Each wave: 32x64 output tile (one head-dim span). 4 waves / block.
// qb,kb: [bh][T][64]   vt: [bh][64][T]
// ---------------------------------------------------------------------------
__global__ void __launch_bounds__(128)
qkv_rope_kernel(const bf16* __restrict__ xb, const bf16* __restrict__ wqkvT,
                bf16* __restrict__ qb, bf16* __restrict__ kb, bf16* __restrict__ vt) {
    const int lane = threadIdx.x & 31;
    const int warp = threadIdx.x >> 5;
    const int tile = blockIdx.x * 4 + warp;       // 6144 tiles
    const int mt = tile / 48, nt = tile - mt * 48;
    const int m0 = mt << 5, n0 = nt << 6;

    const v8f zero = {0.f,0.f,0.f,0.f,0.f,0.f,0.f,0.f};
    v8f c[2][4];
    #pragma unroll
    for (int mi = 0; mi < 2; ++mi)
        #pragma unroll
        for (int f = 0; f < 4; ++f) c[mi][f] = zero;

    for (int k0 = 0; k0 < D; k0 += 32) {
        v16bf a0 = load_afrag(xb, D, m0,      k0, lane);
        v16bf a1 = load_afrag(xb, D, m0 + 16, k0, lane);
        #pragma unroll
        for (int f = 0; f < 4; ++f) {
            v16bf b = load_bfrag(wqkvT, D, n0 + f * 16, k0, lane);
            c[0][f] = wmma_bf16(a0, b, c[0][f]);
            c[1][f] = wmma_bf16(a1, b, c[1][f]);
        }
    }

    const int col = lane & 15;
    const int sec = n0 >> 10;                 // 0=Q 1=K 2=V
    const int h   = (n0 & 1023) >> 6;

    #pragma unroll
    for (int mi = 0; mi < 2; ++mi) {
        const int rbase = m0 + mi * 16 + ((lane >> 4) << 3);
        if (sec < 2) {
            bf16* dstbuf = (sec == 0) ? qb : kb;
            const float qscale = (sec == 0) ? 0.125f : 1.0f;  // 1/sqrt(64) into Q
            #pragma unroll
            for (int fp = 0; fp < 2; ++fp) {
                const int j = fp * 16 + col;                  // rotation index 0..31
                const float theta = __powf(10000.0f, -(float)j * (1.0f / 32.0f));
                #pragma unroll
                for (int i = 0; i < 8; ++i) {
                    const int r = rbase + i;
                    const int b = r >> 11, t = r & 2047;
                    float sn, cs;
                    __sincosf((float)t * theta, &sn, &cs);
                    const float lo = c[mi][fp][i], hi = c[mi][fp + 2][i];
                    const float nlo = (lo * cs - hi * sn) * qscale;
                    const float nhi = (hi * cs + lo * sn) * qscale;
                    bf16* dst = dstbuf + ((((size_t)(b * H + h)) * T + t) << 6);
                    dst[fp * 16 + col]       = (bf16)nlo;
                    dst[(fp + 2) * 16 + col] = (bf16)nhi;
                }
            }
        } else {
            #pragma unroll
            for (int f = 0; f < 4; ++f)
                #pragma unroll
                for (int i = 0; i < 8; ++i) {
                    const int r = rbase + i;
                    const int b = r >> 11, t = r & 2047;
                    const int d = f * 16 + col;
                    vt[((((size_t)(b * H + h)) * HD + d) << 11) + t] = (bf16)c[mi][f][i];
                }
        }
    }
}

// ---------------------------------------------------------------------------
// Stage 2: causal flash-attention. One wave per (bh, 32-query tile).
// K/V fragments loaded once per 32-key chunk, reused by both query halves.
// ---------------------------------------------------------------------------
__global__ void __launch_bounds__(32)
flash_attn_kernel(const bf16* __restrict__ qb, const bf16* __restrict__ kbuf,
                  const bf16* __restrict__ vt, bf16* __restrict__ attn) {
    __shared__ __align__(32) bf16 pl[32 * 32];

    const int lane = threadIdx.x;
    const int bh = blockIdx.x >> 6;         // 0..31
    const int q0 = (blockIdx.x & 63) << 5;  // 32-query tile

    const bf16* Q = qb   + (size_t)bh * T * HD;
    const bf16* K = kbuf + (size_t)bh * T * HD;
    const bf16* V = vt   + (size_t)bh * HD * T;

    v16bf aq[2][2];
    #pragma unroll
    for (int qi = 0; qi < 2; ++qi) {
        aq[qi][0] = load_afrag(Q, HD, q0 + qi * 16, 0,  lane);
        aq[qi][1] = load_afrag(Q, HD, q0 + qi * 16, 32, lane);
    }

    const v8f zero = {0.f,0.f,0.f,0.f,0.f,0.f,0.f,0.f};
    v8f acc[2][4];
    float mrow[2][8], lrow[2][8];
    #pragma unroll
    for (int qi = 0; qi < 2; ++qi) {
        #pragma unroll
        for (int f = 0; f < 4; ++f) acc[qi][f] = zero;
        #pragma unroll
        for (int i = 0; i < 8; ++i) { mrow[qi][i] = -1e30f; lrow[qi][i] = 0.f; }
    }

    const int col   = lane & 15;
    const int rhalf = (lane >> 4) << 3;

    for (int kc = 0; kc < q0 + 32; kc += 32) {
        // K fragments for this chunk (shared by both query halves)
        v16bf b00 = load_bfrag(K, HD, kc,      0,  lane);
        v16bf b01 = load_bfrag(K, HD, kc,      32, lane);
        v16bf b10 = load_bfrag(K, HD, kc + 16, 0,  lane);
        v16bf b11 = load_bfrag(K, HD, kc + 16, 32, lane);

        #pragma unroll
        for (int qi = 0; qi < 2; ++qi) {
            v8f s0 = zero, s1 = zero;
            s0 = wmma_bf16(aq[qi][0], b00, s0);
            s0 = wmma_bf16(aq[qi][1], b01, s0);
            s1 = wmma_bf16(aq[qi][0], b10, s1);
            s1 = wmma_bf16(aq[qi][1], b11, s1);

            const int row0 = q0 + qi * 16;
            if (kc + 32 > row0) {   // causal mask touches this chunk (uniform)
                #pragma unroll
                for (int i = 0; i < 8; ++i) {
                    const int t = row0 + rhalf + i;
                    if (kc + col > t)      s0[i] = -1e30f;
                    if (kc + 16 + col > t) s1[i] = -1e30f;
                }
            }
            #pragma unroll
            for (int i = 0; i < 8; ++i) {
                float v = rmax16(fmaxf(s0[i], s1[i]));
                const float mnew  = fmaxf(mrow[qi][i], v);
                const float scale = __expf(mrow[qi][i] - mnew);
                const float e0 = __expf(s0[i] - mnew);
                const float e1 = __expf(s1[i] - mnew);
                const float rs = rsum16(e0 + e1);
                lrow[qi][i] = lrow[qi][i] * scale + rs;
                mrow[qi][i] = mnew;
                #pragma unroll
                for (int f = 0; f < 4; ++f) acc[qi][f][i] *= scale;
                // P tile (C-layout) -> LDS, row = 16*qi + local row
                pl[(qi * 16 + rhalf + i) * 32 + col]      = (bf16)e0;
                pl[(qi * 16 + rhalf + i) * 32 + 16 + col] = (bf16)e1;
            }
        }
        __syncthreads();
        v16bf ap0 = load_afrag(pl, 32, 0,  0, lane);
        v16bf ap1 = load_afrag(pl, 32, 16, 0, lane);
        __syncthreads();
        #pragma unroll
        for (int f = 0; f < 4; ++f) {
            v16bf bv = load_bfrag(V, T, f * 16, kc, lane);  // once per chunk
            acc[0][f] = wmma_bf16(ap0, bv, acc[0][f]);
            acc[1][f] = wmma_bf16(ap1, bv, acc[1][f]);
        }
    }

    const int b = bh >> 4, h = bh & 15;
    #pragma unroll
    for (int qi = 0; qi < 2; ++qi)
        #pragma unroll
        for (int f = 0; f < 4; ++f)
            #pragma unroll
            for (int i = 0; i < 8; ++i) {
                const int t = q0 + qi * 16 + rhalf + i;
                const int d = f * 16 + col;
                attn[((size_t)b * T + t) * D + h * HD + d] =
                    (bf16)(acc[qi][f][i] / lrow[qi][i]);
            }
}

// ---------------------------------------------------------------------------
// Stage 3: output projection + bias (fp32 result), 32x64 tiles per wave
// ---------------------------------------------------------------------------
__global__ void __launch_bounds__(128)
out_proj_kernel(const bf16* __restrict__ attn, const bf16* __restrict__ woutT,
                const float* __restrict__ bias, float* __restrict__ out) {
    const int lane = threadIdx.x & 31;
    const int warp = threadIdx.x >> 5;
    const int tile = blockIdx.x * 4 + warp;       // 2048 tiles
    const int mt = tile >> 4, nt = tile & 15;
    const int m0 = mt << 5, n0 = nt << 6;

    const v8f zero = {0.f,0.f,0.f,0.f,0.f,0.f,0.f,0.f};
    v8f c[2][4];
    #pragma unroll
    for (int mi = 0; mi < 2; ++mi)
        #pragma unroll
        for (int f = 0; f < 4; ++f) c[mi][f] = zero;

    for (int k0 = 0; k0 < D; k0 += 32) {
        v16bf a0 = load_afrag(attn, D, m0,      k0, lane);
        v16bf a1 = load_afrag(attn, D, m0 + 16, k0, lane);
        #pragma unroll
        for (int f = 0; f < 4; ++f) {
            v16bf b = load_bfrag(woutT, D, n0 + f * 16, k0, lane);
            c[0][f] = wmma_bf16(a0, b, c[0][f]);
            c[1][f] = wmma_bf16(a1, b, c[1][f]);
        }
    }

    const int col = lane & 15;
    #pragma unroll
    for (int mi = 0; mi < 2; ++mi) {
        const int rbase = m0 + mi * 16 + ((lane >> 4) << 3);
        #pragma unroll
        for (int f = 0; f < 4; ++f)
            #pragma unroll
            for (int i = 0; i < 8; ++i) {
                const int r = rbase + i;
                const int n = n0 + f * 16 + col;
                out[(size_t)r * D + n] = c[mi][f][i] + bias[n];
            }
    }
}

// ---------------------------------------------------------------------------
// Launcher
// ---------------------------------------------------------------------------
extern "C" void kernel_launch(void* const* d_in, const int* in_sizes, int n_in,
                              void* d_out, int out_size, void* d_ws, size_t ws_size,
                              hipStream_t stream) {
    const float* x     = (const float*)d_in[0];   // [2,2048,1024]
    const float* w_qkv = (const float*)d_in[1];   // [1024,3072]
    const float* w_out = (const float*)d_in[2];   // [1024,1024]
    const float* b_out = (const float*)d_in[3];   // [1024]
    float* out = (float*)d_out;

    char* ws = (char*)d_ws;
    const size_t MiB = (size_t)1 << 20;
    bf16* xb    = (bf16*)(ws + 0 * MiB);   // 8 MiB   x in bf16 [4096][1024]
    bf16* wqkvT = (bf16*)(ws + 8 * MiB);   // 6 MiB   [3072][1024]
    bf16* woutT = (bf16*)(ws + 14 * MiB);  // 2 MiB   [1024][1024]
    bf16* qb    = (bf16*)(ws + 16 * MiB);  // 8 MiB   [32][2048][64]
    bf16* kb    = (bf16*)(ws + 24 * MiB);  // 8 MiB   [32][2048][64]
    bf16* vt    = (bf16*)(ws + 32 * MiB);  // 8 MiB   [32][64][2048]
    bf16* attn  = (bf16*)(ws + 40 * MiB);  // 8 MiB   [4096][1024]

    cvt_f32_to_bf16<<<(M * D) / 256, 256, 0, stream>>>(x, xb, M * D);
    transpose_to_bf16<<<(D * 3 * D) / 256, 256, 0, stream>>>(w_qkv, wqkvT, D, 3 * D);
    transpose_to_bf16<<<(D * D) / 256, 256, 0, stream>>>(w_out, woutT, D, D);

    qkv_rope_kernel<<<1536, 128, 0, stream>>>(xb, wqkvT, qb, kb, vt);
    flash_attn_kernel<<<2048, 32, 0, stream>>>(qb, kb, vt, attn);
    out_proj_kernel<<<512, 128, 0, stream>>>(attn, woutT, b_out, out);
}
